// LinearAttention_79517024518809
// MI455X (gfx1250) — matrix operand order repaired and checked by reading, not compiled, not run on hardware
//
#include <hip/hip_runtime.h>
#include <hip/hip_bf16.h>

typedef __attribute__((ext_vector_type(16))) __bf16 v16bf;
typedef __attribute__((ext_vector_type(8)))  float  v8f;

#define BATCH 16
#define CH 256
#define NPIX 4096
#define HEADS 8
#define HD 32
#define QKVROWS 768
#define NMEM 4
#define QSCALE 0.17677669529663687f /* 32^-0.5 */
#define LN_EPS 1e-5f

// ---------------- workspace layout (bytes) ----------------
static const size_t WS_WQ   = 0;                                   // 768*256 bf16
static const size_t WS_WO   = WS_WQ   + (size_t)768 * 256 * 2;     // 256*256 bf16
static const size_t WS_KMAX = WS_WO   + (size_t)256 * 256 * 2;     // B*256 uint (ordered-float encoding)
static const size_t WS_CTX  = WS_KMAX + (size_t)BATCH * 256 * 4;   // B*8*32*32 f32
static const size_t WS_Q    = WS_CTX  + (size_t)BATCH * HEADS * HD * HD * 4;
static const size_t WS_K    = WS_Q + (size_t)BATCH * CH * NPIX * 4;
static const size_t WS_V    = WS_K + (size_t)BATCH * CH * NPIX * 4;

// ordered-uint encoding so we can use integer atomicMax on floats
__device__ __forceinline__ unsigned fenc(float f) {
  unsigned u = __float_as_uint(f);
  return (u & 0x80000000u) ? ~u : (u | 0x80000000u);
}
__device__ __forceinline__ float fdec(unsigned k) {
  unsigned u = (k & 0x80000000u) ? (k ^ 0x80000000u) : ~k;
  return __uint_as_float(u);
}

union FragB16 { v16bf v; uint4 q[2]; };

// A fragment, 16-bit 16x32: lane holds K = kbase..kbase+15
__device__ __forceinline__ v16bf load_frag_a_bf16(const __bf16* row, int kbase) {
  FragB16 f;
  f.q[0] = *(const uint4*)(row + kbase);
  f.q[1] = *(const uint4*)(row + kbase + 8);
  return f.v;
}
// B fragment, 16-bit 32x16: lane holds 16 consecutive K for its column
__device__ __forceinline__ v16bf load_frag_b_bf16(const __bf16* col) {
  FragB16 f;
  f.q[0] = *(const uint4*)(col);
  f.q[1] = *(const uint4*)(col + 8);
  return f.v;
}
__device__ __forceinline__ v8f wmma_bf16(v16bf a, v16bf b, v8f c) {
  return __builtin_amdgcn_wmma_f32_16x16x32_bf16(false, a, false, b, (short)0, c,
                                                 false, false);
}

// ---------------- kernel 0: weight cast + kmax init ----------------
__global__ void k_prep(const float* __restrict__ wq, const float* __restrict__ wo,
                       __bf16* __restrict__ wqb, __bf16* __restrict__ wob,
                       unsigned* __restrict__ kmax) {
  int idx = blockIdx.x * blockDim.x + threadIdx.x;
  if (idx < 768 * 256) wqb[idx] = (__bf16)wq[idx];
  if (idx < 256 * 256) wob[idx] = (__bf16)wo[idx];
  if (idx < BATCH * 256) kmax[idx] = 0u; // minimum ordered key
}

// ---------------- kernel 1: LN + QKV GEMM + q-softmax + k-max ----------------
// grid (128 pixel tiles, 16 batches), block 256 (8 waves)
__global__ void k1_ln_qkv(const float* __restrict__ x, const float* __restrict__ norm_g,
                          const __bf16* __restrict__ wqb,
                          float* __restrict__ qbuf, float* __restrict__ kbuf,
                          float* __restrict__ vbuf, unsigned* __restrict__ kmax) {
  extern __shared__ char smem[];
  // region0: xs[32][257] f32 (raw tile) overlapped with qkvs[768][33] f32 (results)
  float*  xs   = (float*)smem;
  float*  qkvs = (float*)smem;
  __bf16* xnb  = (__bf16*)(smem + (size_t)768 * 33 * 4);        // [32][264] bf16
  float*  red1 = (float*)((char*)xnb + (size_t)32 * 264 * 2);   // [32][8]
  float*  red2 = red1 + 256;                                    // [32][8]
  float*  smean = red2 + 256;                                   // [32]
  float*  srstd = smean + 32;                                   // [32]

  const int t = threadIdx.x;
  const int bb = blockIdx.y;
  const int n0 = blockIdx.x * 32;
  const int lane = t & 31, wv = t >> 5;
  const int mrow = lane & 15, hi = lane >> 4;

  // ---- load x tile into xs[n][c] (coalesced over n) ----
  {
    const int n = t & 31, cs = t >> 5;
    for (int c0 = 0; c0 < CH; c0 += 8) {
      int c = c0 + cs;
      xs[n * 257 + c] = x[((size_t)bb * CH + c) * NPIX + n0 + n];
    }
  }
  __syncthreads();
  // ---- per-pixel channel stats (8 threads / pixel) ----
  {
    int p = t >> 3, part = t & 7;
    float s1 = 0.f, s2 = 0.f;
    for (int i = 0; i < 32; ++i) {
      float v = xs[p * 257 + part * 32 + i];
      s1 += v; s2 += v * v;
    }
    red1[p * 8 + part] = s1;
    red2[p * 8 + part] = s2;
  }
  __syncthreads();
  if (t < 32) {
    float s1 = 0.f, s2 = 0.f;
    for (int i = 0; i < 8; ++i) { s1 += red1[t * 8 + i]; s2 += red2[t * 8 + i]; }
    float mean = s1 * (1.f / CH);
    float var  = s2 * (1.f / CH) - mean * mean;
    smean[t] = mean;
    srstd[t] = rsqrtf(var + LN_EPS);
  }
  __syncthreads();
  // ---- normalized bf16 tile xnb[n][c] ----
  {
    const int n = t & 31, cs = t >> 5;
    float mean = smean[n], rstd = srstd[n];
    for (int c0 = 0; c0 < CH; c0 += 8) {
      int c = c0 + cs;
      xnb[n * 264 + c] = (__bf16)((xs[n * 257 + c] - mean) * rstd * norm_g[c]);
    }
  }
  __syncthreads();

  // ---- GEMM: qkv(768x32) = Wqkv(768x256) x xnb(256x32); wave -> 96 rows ----
  {
    const int m0 = wv * 96;
    v8f acc[6][2];
    v8f z = {};
    for (int i = 0; i < 6; ++i) for (int j = 0; j < 2; ++j) acc[i][j] = z;
    for (int kk = 0; kk < CH; kk += 32) {
      v16bf bfrag[2];
      for (int j = 0; j < 2; ++j) {
        int n = j * 16 + mrow;
        bfrag[j] = load_frag_b_bf16(xnb + n * 264 + kk + hi * 16);
      }
      for (int i = 0; i < 6; ++i) {
        v16bf a = load_frag_a_bf16(wqb + (size_t)(m0 + i * 16 + mrow) * CH + kk, hi * 16);
        for (int j = 0; j < 2; ++j)
          acc[i][j] = wmma_bf16(a, bfrag[j], acc[i][j]);
      }
    }
    for (int i = 0; i < 6; ++i)
      for (int j = 0; j < 2; ++j)
        for (int r = 0; r < 8; ++r) {
          int row = m0 + i * 16 + r + hi * 8;
          int col = j * 16 + mrow;
          qkvs[row * 33 + col] = acc[i][j][r];
        }
  }
  __syncthreads();

  // ---- q: softmax over head_dim, scaled. k: per-(h,d) max -> atomicMax ----
  {
    int h = t >> 5, p = t & 31;
    float mx = -1e30f;
    for (int d = 0; d < HD; ++d) mx = fmaxf(mx, qkvs[(h * 32 + d) * 33 + p]);
    float s = 0.f;
    for (int d = 0; d < HD; ++d) s += __expf(qkvs[(h * 32 + d) * 33 + p] - mx);
    float inv = QSCALE / s;
    for (int d = 0; d < HD; ++d) {
      int idx = (h * 32 + d) * 33 + p;
      qkvs[idx] = __expf(qkvs[idx] - mx) * inv;
    }
    float km = -1e30f;
    for (int n = 0; n < 32; ++n) km = fmaxf(km, qkvs[(256 + t) * 33 + n]);
    atomicMax(&kmax[bb * 256 + t], fenc(km));
  }
  __syncthreads();

  // ---- coalesced writeback of q (softmaxed), k, v ----
  {
    int n = t & 31, rs = t >> 5;
    for (int r0 = 0; r0 < QKVROWS; r0 += 8) {
      int row = r0 + rs;
      float v = qkvs[row * 33 + n];
      if (row < 256)      qbuf[((size_t)bb * CH + row)       * NPIX + n0 + n] = v;
      else if (row < 512) kbuf[((size_t)bb * CH + (row-256)) * NPIX + n0 + n] = v;
      else                vbuf[((size_t)bb * CH + (row-512)) * NPIX + n0 + n] = v;
    }
  }
}

// ---------------- kernel 2: ctx = softmax_tokens(k) . v^T  (per (b,h)) ----------------
// grid (8 heads, 16 batches), block 256 (8 waves, each 512 tokens)
__global__ void k2_ctx(const float* __restrict__ kbuf, const float* __restrict__ vbuf,
                       const unsigned* __restrict__ kmax, const float* __restrict__ mem_kv,
                       float* __restrict__ ctxbuf) {
  __shared__ float smaxd[32];
  __shared__ float sS[32];
  __shared__ float sctx[32][32];
  const int t = threadIdx.x;
  const int h = blockIdx.x, bb = blockIdx.y;
  const int lane = t & 31, wv = t >> 5;
  const int mrow = lane & 15, hi = lane >> 4;

  if (t < 32) {
    float m = fdec(kmax[bb * 256 + h * 32 + t]);
    for (int j = 0; j < NMEM; ++j)
      m = fmaxf(m, mem_kv[((size_t)(0 * HEADS + h) * HD + t) * NMEM + j]);
    smaxd[t] = m;
    sS[t] = 0.f;
  }
  for (int i = t; i < 1024; i += 256) ((float*)sctx)[i] = 0.f;
  __syncthreads();

  v8f acc[2][2];
  v8f z = {};
  for (int i = 0; i < 2; ++i) for (int j = 0; j < 2; ++j) acc[i][j] = z;
  float spart[2] = {0.f, 0.f};
  const float mx0 = smaxd[mrow], mx1 = smaxd[16 + mrow];

  for (int ch = 0; ch < 16; ++ch) {
    int n0 = wv * 512 + ch * 32;
    v16bf afrag[2];
    for (int ti = 0; ti < 2; ++ti) {
      int d = ti * 16 + mrow;
      const float* kr = kbuf + ((size_t)bb * 256 + h * 32 + d) * NPIX + n0 + hi * 16;
      float kv[16];
      *(float4*)&kv[0]  = *(const float4*)(kr);
      *(float4*)&kv[4]  = *(const float4*)(kr + 4);
      *(float4*)&kv[8]  = *(const float4*)(kr + 8);
      *(float4*)&kv[12] = *(const float4*)(kr + 12);
      float m = (ti == 0) ? mx0 : mx1;
      FragB16 f;
      float s = 0.f;
      for (int i2 = 0; i2 < 16; ++i2) {
        float e = __expf(kv[i2] - m);
        s += e;
        f.v[i2] = (__bf16)e;
      }
      spart[ti] += s;
      afrag[ti] = f.v;
    }
    v16bf bfrag[2];
    for (int tj = 0; tj < 2; ++tj) {
      int e = tj * 16 + mrow;
      const float* vr = vbuf + ((size_t)bb * 256 + h * 32 + e) * NPIX + n0 + hi * 16;
      float vv[16];
      *(float4*)&vv[0]  = *(const float4*)(vr);
      *(float4*)&vv[4]  = *(const float4*)(vr + 4);
      *(float4*)&vv[8]  = *(const float4*)(vr + 8);
      *(float4*)&vv[12] = *(const float4*)(vr + 12);
      FragB16 f;
      for (int i2 = 0; i2 < 16; ++i2) f.v[i2] = (__bf16)vv[i2];
      bfrag[tj] = f.v;
    }
    for (int ti = 0; ti < 2; ++ti)
      for (int tj = 0; tj < 2; ++tj)
        acc[ti][tj] = wmma_bf16(afrag[ti], bfrag[tj], acc[ti][tj]);
  }

  atomicAdd(&sS[mrow], spart[0]);
  atomicAdd(&sS[16 + mrow], spart[1]);
  for (int ti = 0; ti < 2; ++ti)
    for (int tj = 0; tj < 2; ++tj)
      for (int r = 0; r < 8; ++r) {
        int d = ti * 16 + r + hi * 8;
        int e = tj * 16 + mrow;
        atomicAdd(&sctx[d][e], acc[ti][tj][r]);
      }
  __syncthreads();

  if (t < 32) {
    int d = t;
    float m = smaxd[d];
    float em[NMEM];
    float S = sS[d];
    for (int j = 0; j < NMEM; ++j) {
      em[j] = __expf(mem_kv[((size_t)(0 * HEADS + h) * HD + d) * NMEM + j] - m);
      S += em[j];
    }
    float invS = 1.f / S;
    for (int e = 0; e < 32; ++e) {
      float c = sctx[d][e];
      for (int j = 0; j < NMEM; ++j)
        c += em[j] * mem_kv[((size_t)(1 * HEADS + h) * HD + e) * NMEM + j];
      ctxbuf[(((size_t)bb * HEADS + h) * HD + d) * HD + e] = c * invS;
    }
  }
}

// ---------------- kernel 3: out = LN( Wout . (ctx^T q) + b ) ----------------
// grid (128 pixel tiles, 16 batches), block 256 (8 waves: stepA wave==head)
__global__ void k3_out(const float* __restrict__ qbuf, const float* __restrict__ ctxbuf,
                       const __bf16* __restrict__ wob, const float* __restrict__ b_out,
                       const float* __restrict__ out_g, float* __restrict__ out) {
  extern __shared__ char smem[];
  float*  sctx  = (float*)smem;                                   // [8][32][32]
  __bf16* sattn = (__bf16*)(smem + (size_t)8 * 32 * 32 * 4);      // [32][264]
  float*  sy    = (float*)((char*)sattn + (size_t)32 * 264 * 2);  // [256][33]
  float*  red1  = sy + 256 * 33;
  float*  red2  = red1 + 256;
  float*  smean = red2 + 256;
  float*  srstd = smean + 32;

  const int t = threadIdx.x;
  const int bb = blockIdx.y;
  const int n0 = blockIdx.x * 32;
  const int lane = t & 31, wv = t >> 5;
  const int mrow = lane & 15, hi = lane >> 4;

  for (int i = t; i < 8 * 32 * 32; i += 256)
    sctx[i] = ctxbuf[(size_t)bb * 8 * 32 * 32 + i];
  __syncthreads();

  // ---- step A: per-head attn[e][n] = sum_d ctx[d][e] q[d][n]; wave = head ----
  {
    const int h = wv;
    v16bf afrag[2];
    for (int ti = 0; ti < 2; ++ti) {
      int e = ti * 16 + mrow;
      FragB16 f;
      for (int i2 = 0; i2 < 16; ++i2) {
        int d = hi * 16 + i2;
        f.v[i2] = (__bf16)sctx[(h * 32 + d) * 32 + e];
      }
      afrag[ti] = f.v;
    }
    v16bf bfrag[2];
    for (int tj = 0; tj < 2; ++tj) {
      int n = tj * 16 + mrow;
      FragB16 f;
      for (int i2 = 0; i2 < 16; ++i2) {
        int d = hi * 16 + i2;
        f.v[i2] = (__bf16)qbuf[((size_t)bb * CH + h * 32 + d) * NPIX + n0 + n];
      }
      bfrag[tj] = f.v;
    }
    v8f z = {};
    for (int ti = 0; ti < 2; ++ti)
      for (int tj = 0; tj < 2; ++tj) {
        v8f dacc = wmma_bf16(afrag[ti], bfrag[tj], z);
        for (int r = 0; r < 8; ++r) {
          int e = ti * 16 + r + hi * 8;
          int n = tj * 16 + mrow;
          sattn[n * 264 + h * 32 + e] = (__bf16)dacc[r];
        }
      }
  }
  __syncthreads();

  // ---- step B: y(256x32) = Wout(256x256) x attn; wave -> 32 rows ----
  {
    const int m0 = wv * 32;
    v8f acc[2][2];
    v8f z = {};
    for (int i = 0; i < 2; ++i) for (int j = 0; j < 2; ++j) acc[i][j] = z;
    for (int kk = 0; kk < CH; kk += 32) {
      v16bf bfrag[2];
      for (int j = 0; j < 2; ++j) {
        int n = j * 16 + mrow;
        bfrag[j] = load_frag_b_bf16(sattn + n * 264 + kk + hi * 16);
      }
      for (int i = 0; i < 2; ++i) {
        v16bf a = load_frag_a_bf16(wob + (size_t)(m0 + i * 16 + mrow) * CH + kk, hi * 16);
        for (int j = 0; j < 2; ++j) acc[i][j] = wmma_bf16(a, bfrag[j], acc[i][j]);
      }
    }
    for (int i = 0; i < 2; ++i)
      for (int j = 0; j < 2; ++j)
        for (int r = 0; r < 8; ++r) {
          int row = m0 + i * 16 + r + hi * 8;
          int col = j * 16 + mrow;
          sy[row * 33 + col] = acc[i][j][r] + b_out[row];
        }
  }
  __syncthreads();

  // ---- step C: per-pixel channel layernorm ----
  {
    int p = t >> 3, part = t & 7;
    float s1 = 0.f, s2 = 0.f;
    for (int i = 0; i < 32; ++i) {
      float v = sy[(part * 32 + i) * 33 + p];
      s1 += v; s2 += v * v;
    }
    red1[p * 8 + part] = s1;
    red2[p * 8 + part] = s2;
  }
  __syncthreads();
  if (t < 32) {
    float s1 = 0.f, s2 = 0.f;
    for (int i = 0; i < 8; ++i) { s1 += red1[t * 8 + i]; s2 += red2[t * 8 + i]; }
    float mean = s1 * (1.f / CH);
    float var  = s2 * (1.f / CH) - mean * mean;
    smean[t] = mean;
    srstd[t] = rsqrtf(var + LN_EPS);
  }
  __syncthreads();
  {
    int n = t & 31, rs = t >> 5;
    float mean = smean[n], rstd = srstd[n];
    for (int r0 = 0; r0 < CH; r0 += 8) {
      int c = r0 + rs;
      out[((size_t)bb * CH + c) * NPIX + n0 + n] = (sy[c * 33 + n] - mean) * rstd * out_g[c];
    }
  }
}

extern "C" void kernel_launch(void* const* d_in, const int* in_sizes, int n_in,
                              void* d_out, int out_size, void* d_ws, size_t ws_size,
                              hipStream_t stream) {
  (void)in_sizes; (void)n_in; (void)out_size; (void)ws_size;
  const float* x      = (const float*)d_in[0];
  const float* norm_g = (const float*)d_in[1];
  const float* w_qkv  = (const float*)d_in[2];
  const float* mem_kv = (const float*)d_in[3];
  const float* w_out  = (const float*)d_in[4];
  const float* b_out  = (const float*)d_in[5];
  const float* out_g  = (const float*)d_in[6];

  char* ws = (char*)d_ws;
  __bf16*   wqb  = (__bf16*)(ws + WS_WQ);
  __bf16*   wob  = (__bf16*)(ws + WS_WO);
  unsigned* kmax = (unsigned*)(ws + WS_KMAX);
  float*    ctx  = (float*)(ws + WS_CTX);
  float*    qb   = (float*)(ws + WS_Q);
  float*    kb   = (float*)(ws + WS_K);
  float*    vb   = (float*)(ws + WS_V);
  float*    outp = (float*)d_out;

  k_prep<<<768, 256, 0, stream>>>(w_qkv, w_out, wqb, wob, kmax);

  size_t lds1 = (size_t)768 * 33 * 4 + (size_t)32 * 264 * 2 + 2 * 256 * 4 + 2 * 32 * 4;
  k1_ln_qkv<<<dim3(128, 16), 256, lds1, stream>>>(x, norm_g, wqb, qb, kb, vb, kmax);

  k2_ctx<<<dim3(8, 16), 256, 0, stream>>>(kb, vb, kmax, mem_kv, ctx);

  size_t lds3 = (size_t)8 * 32 * 32 * 4 + (size_t)32 * 264 * 2 + (size_t)256 * 33 * 4 +
                2 * 256 * 4 + 2 * 32 * 4;
  k3_out<<<dim3(128, 16), 256, lds3, stream>>>(qb, ctx, wob, b_out, out_g, outp);
}
